// MultiHeadedAttention_24730421690494
// MI455X (gfx1250) — compile-verified, matrix operand order checked by reading, not run on hardware
//
#include <hip/hip_runtime.h>
#include <hip/hip_bf16.h>
#include <cstdint>
#include <cstddef>

// ---------------- types ----------------
typedef unsigned short u16;
typedef unsigned int   u32;
typedef __attribute__((ext_vector_type(16))) __bf16 v16bf;
typedef __attribute__((ext_vector_type(8)))  __bf16 v8bf;
typedef __attribute__((ext_vector_type(8)))  float  v8f;

#define WMMA_BF16(a, b, c) \
  __builtin_amdgcn_wmma_f32_16x16x32_bf16(false, (a), false, (b), (short)0, (c), false, false)

// CDNA5 async LDS-DMA path (GLOBAL_LOAD_ASYNC_TO_LDS_B128, ASYNCcnt).
#if defined(__AMDGCN__) && \
    __has_builtin(__builtin_amdgcn_global_load_async_to_lds_b128) && \
    __has_builtin(__builtin_amdgcn_s_wait_asynccnt)
#define USE_ASYNC 1
#else
#define USE_ASYNC 0
#endif

#if USE_ASYNC
// ROCm 7.2 clang signature: (int4 AS1*, int4 AS3*, Imm offset, Imm cpol)
typedef int i32x4 __attribute__((vector_size(16)));
typedef __attribute__((address_space(1))) i32x4 g_i32x4_t;
typedef __attribute__((address_space(3))) i32x4 l_i32x4_t;
static __device__ __forceinline__ void async_cp16(void* lds, const void* g) {
  __builtin_amdgcn_global_load_async_to_lds_b128(
      (g_i32x4_t*)g, (l_i32x4_t*)lds, /*offset=*/0, /*cpol=*/0);
}
static __device__ __forceinline__ void wait_async() {
  __builtin_amdgcn_s_wait_asynccnt(0);
}
#else
static __device__ __forceinline__ void wait_async() {}
#endif

// problem constants
constexpr int BB = 2;
constexpr int S  = 2048;
constexpr int D  = 1024;
constexpr int H  = 16;
constexpr int N  = BB * S;          // 4096 rows

static __device__ __forceinline__ u16 f2bf_rne(float f) {
  u32 u = __float_as_uint(f);
  u += 0x7fffu + ((u >> 16) & 1u);  // round-to-nearest-even
  return (u16)(u >> 16);
}

// Build a v16bf fragment from two 16-byte chunks (A/B fragment layout:
// lane holds K in [hi*8, hi*8+8) and [16+hi*8, 16+hi*8+8) of a 32-wide K slice).
static __device__ __forceinline__ v16bf make_frag(const u16* p0, const u16* p1) {
  v8bf lo = *(const v8bf*)p0;
  v8bf hi = *(const v8bf*)p1;
  v16bf r;
#pragma unroll
  for (int i = 0; i < 8; ++i) { r[i] = lo[i]; r[i + 8] = hi[i]; }
  return r;
}

// ---------------- fp32 -> bf16 convert ----------------
__global__ __launch_bounds__(256) void cvt_f32_bf16(const float* __restrict__ src,
                                                    u16* __restrict__ dst, int n) {
  int i = (blockIdx.x * 256 + threadIdx.x) * 4;
  if (i + 3 < n) {
    float4 f = *(const float4*)(src + i);
    ushort4 o;
    o.x = f2bf_rne(f.x); o.y = f2bf_rne(f.y);
    o.z = f2bf_rne(f.z); o.w = f2bf_rne(f.w);
    *(ushort4*)(dst + i) = o;
  }
}

// ---------------- QKV projection GEMM ----------------
// Y = X @ W^T  (X: [N,D] bf16 row-major, W: [D,D] bf16 row-major => B-frag reads W rows)
// z = blockIdx.z selects Q(0)/K(1)/V(2). Epilogue: split heads; Q scaled 1/sqrt(64);
// V stored transposed per head: Vt[b][h][d][s].
__global__ __launch_bounds__(256) void qkv_gemm(
    const u16* __restrict__ X,
    const u16* __restrict__ Wq, const u16* __restrict__ Wk, const u16* __restrict__ Wv,
    u16* __restrict__ Qh, u16* __restrict__ Kh, u16* __restrict__ Vt) {
  // double-buffered 128x32 tiles, stride 40 (bank-conflict-free fragment reads)
  __shared__ alignas(16) u16 As[2][128][40];
  __shared__ alignas(16) u16 Bs[2][128][40];

  const int z = blockIdx.z;
  const u16* W = (z == 0) ? Wq : ((z == 1) ? Wk : Wv);
  const int rowBase = blockIdx.x * 128;   // over N
  const int colBase = blockIdx.y * 128;   // over D
  const int tid = threadIdx.x, lane = tid & 31, wave = tid >> 5;
  const int wm = wave & 3, wn = wave >> 2;      // 4 (M) x 2 (N) waves
  const int l16 = lane & 15, hi = lane >> 4;

  auto issue_tile = [&](int buf, int k0) {
#if USE_ASYNC
#pragma unroll
    for (int i = 0; i < 2; ++i) {           // 512 vec8 per operand / 256 thr
      int v  = tid + i * 256;
      int r  = v >> 2;
      int c8 = (v & 3) * 8;
      async_cp16(&As[buf][r][c8], &X[(size_t)(rowBase + r) * D + k0 + c8]);
      async_cp16(&Bs[buf][r][c8], &W[(size_t)(colBase + r) * D + k0 + c8]);
    }
#else
    uint4 ra[2], rb[2];
#pragma unroll
    for (int i = 0; i < 2; ++i) {
      int v  = tid + i * 256;
      int r  = v >> 2;
      int c8 = (v & 3) * 8;
      ra[i] = *(const uint4*)&X[(size_t)(rowBase + r) * D + k0 + c8];
      rb[i] = *(const uint4*)&W[(size_t)(colBase + r) * D + k0 + c8];
    }
#pragma unroll
    for (int i = 0; i < 2; ++i) {
      int v  = tid + i * 256;
      int r  = v >> 2;
      int c8 = (v & 3) * 8;
      *(uint4*)&As[buf][r][c8] = ra[i];
      *(uint4*)&Bs[buf][r][c8] = rb[i];
    }
#endif
  };

  v8f acc[2][4];
  const v8f vzero = {0.f, 0.f, 0.f, 0.f, 0.f, 0.f, 0.f, 0.f};
#pragma unroll
  for (int i = 0; i < 2; ++i)
#pragma unroll
    for (int j = 0; j < 4; ++j) acc[i][j] = vzero;

  issue_tile(0, 0);
  for (int k0 = 0, j = 0; k0 < D; k0 += 32, ++j) {
    const int buf = j & 1;
    wait_async();
    __syncthreads();
    if (k0 + 32 < D) issue_tile(buf ^ 1, k0 + 32);

    const int c0 = hi * 8;
    v16bf a[2], b[4];
#pragma unroll
    for (int mt = 0; mt < 2; ++mt) {
      int r = wm * 32 + mt * 16 + l16;
      a[mt] = make_frag(&As[buf][r][c0], &As[buf][r][c0 + 16]);
    }
#pragma unroll
    for (int nt = 0; nt < 4; ++nt) {
      int r = wn * 64 + nt * 16 + l16;
      b[nt] = make_frag(&Bs[buf][r][c0], &Bs[buf][r][c0 + 16]);
    }
#pragma unroll
    for (int mt = 0; mt < 2; ++mt)
#pragma unroll
      for (int nt = 0; nt < 4; ++nt)
        acc[mt][nt] = WMMA_BF16(a[mt], b[nt], acc[mt][nt]);
  }

  // epilogue: C layout lane (m = r + 8*hi, n = l16) per 16x16 tile
#pragma unroll
  for (int mt = 0; mt < 2; ++mt) {
#pragma unroll
    for (int nt = 0; nt < 4; ++nt) {
#pragma unroll
      for (int r = 0; r < 8; ++r) {
        int row = rowBase + wm * 32 + mt * 16 + r + hi * 8;   // row in [0, N)
        int col = colBase + wn * 64 + nt * 16 + l16;          // col in [0, D)
        int b_  = row >> 11;          // /2048
        int s_  = row & 2047;
        int h_  = col >> 6;
        int d_  = col & 63;
        float v = acc[mt][nt][r];
        if (z == 0) {
          Qh[((size_t)((b_ * H + h_) * S + s_)) * 64 + d_] = f2bf_rne(v * 0.125f);
        } else if (z == 1) {
          Kh[((size_t)((b_ * H + h_) * S + s_)) * 64 + d_] = f2bf_rne(v);
        } else {
          Vt[((size_t)((b_ * H + h_) * 64 + d_)) * S + s_] = f2bf_rne(v);
        }
      }
    }
  }
}

// ---------------- flash attention core ----------------
// One block per (b, h, 64 query rows); 4 waves, each owns 16 query rows.
__global__ __launch_bounds__(128) void attn_kernel(
    const u16* __restrict__ Qh, const u16* __restrict__ Kh, const u16* __restrict__ Vt,
    const unsigned char* __restrict__ mask, u16* __restrict__ Ctx) {
  __shared__ alignas(16) u16 Kt[2][64][72];   // [k-pos][d]
  __shared__ alignas(16) u16 Vs[2][64][72];   // [d][k-pos]   (from transposed V)
  __shared__ alignas(16) u16 Pt[64][72];      // softmaxed probs, bf16

  const int qb = blockIdx.x * 64;
  const int h  = blockIdx.y;
  const int b  = blockIdx.z;
  const int tid = threadIdx.x, lane = tid & 31, wave = tid >> 5;
  const int l16 = lane & 15, hi = lane >> 4;

  const u16* Qbase = Qh + ((size_t)(b * H + h) * S) * 64;
  const u16* Kbase = Kh + ((size_t)(b * H + h) * S) * 64;
  const u16* Vbase = Vt + ((size_t)(b * H + h) * 64) * S;
  const unsigned char* Mb = mask + (size_t)b * S * S;

  auto issue_tile = [&](int buf, int kt) {
#if USE_ASYNC
#pragma unroll
    for (int i = 0; i < 4; ++i) {            // 512 vec8 per operand / 128 thr
      int v  = tid + i * 128;
      int r  = v >> 3;
      int c8 = (v & 7) * 8;
      async_cp16(&Kt[buf][r][c8], &Kbase[(size_t)(kt + r) * 64 + c8]);
      async_cp16(&Vs[buf][r][c8], &Vbase[(size_t)r * S + kt + c8]);
    }
#else
    uint4 rk[4], rv[4];
#pragma unroll
    for (int i = 0; i < 4; ++i) {
      int v  = tid + i * 128;
      int r  = v >> 3;
      int c8 = (v & 7) * 8;
      rk[i] = *(const uint4*)&Kbase[(size_t)(kt + r) * 64 + c8];
      rv[i] = *(const uint4*)&Vbase[(size_t)r * S + kt + c8];
    }
#pragma unroll
    for (int i = 0; i < 4; ++i) {
      int v  = tid + i * 128;
      int r  = v >> 3;
      int c8 = (v & 7) * 8;
      *(uint4*)&Kt[buf][r][c8] = rk[i];
      *(uint4*)&Vs[buf][r][c8] = rv[i];
    }
#endif
  };

  // Q fragments for this wave's 16 rows (2 K-steps of 32 over d_h=64)
  v16bf aq[2];
  {
    const size_t r = (size_t)(qb + wave * 16 + l16) * 64;
#pragma unroll
    for (int ks = 0; ks < 2; ++ks) {
      int c0 = ks * 32 + hi * 8;
      aq[ks] = make_frag(&Qbase[r + c0], &Qbase[r + c0 + 16]);
    }
  }

  v8f accO[4];
  const v8f vzero = {0.f, 0.f, 0.f, 0.f, 0.f, 0.f, 0.f, 0.f};
#pragma unroll
  for (int i = 0; i < 4; ++i) accO[i] = vzero;
  float mrun[8], lrun[8];
#pragma unroll
  for (int r = 0; r < 8; ++r) { mrun[r] = -3.4e38f; lrun[r] = 0.f; }

  issue_tile(0, 0);
  for (int kt = 0, j = 0; kt < S; kt += 64, ++j) {
    const int buf = j & 1;
    wait_async();
    __syncthreads();
    if (kt + 64 < S) issue_tile(buf ^ 1, kt + 64);

    // ---- scores: S = Q K^T (64 k-positions, 4 N tiles) ----
    v8f sc[4];
#pragma unroll
    for (int nt = 0; nt < 4; ++nt) sc[nt] = vzero;
#pragma unroll
    for (int ks = 0; ks < 2; ++ks) {
      const int c0 = ks * 32 + hi * 8;
#pragma unroll
      for (int nt = 0; nt < 4; ++nt) {
        int r = nt * 16 + l16;                 // k-row; K row-major is B^T col-major
        v16bf bk = make_frag(&Kt[buf][r][c0], &Kt[buf][r][c0 + 16]);
        sc[nt] = WMMA_BF16(aq[ks], bk, sc[nt]);
      }
    }

    // ---- mask ----
#pragma unroll
    for (int nt = 0; nt < 4; ++nt) {
      const size_t kcol = (size_t)(kt + nt * 16 + l16);
#pragma unroll
      for (int r = 0; r < 8; ++r) {
        int qrow = qb + wave * 16 + r + hi * 8;
        if (Mb[(size_t)qrow * S + kcol]) sc[nt][r] = -1e18f;
      }
    }

    // ---- online softmax (row shared by 16 lanes of one half-wave) ----
    float alpha[8];
#pragma unroll
    for (int r = 0; r < 8; ++r) {
      float tmx = fmaxf(fmaxf(sc[0][r], sc[1][r]), fmaxf(sc[2][r], sc[3][r]));
#pragma unroll
      for (int off = 1; off < 16; off <<= 1)
        tmx = fmaxf(tmx, __shfl_xor(tmx, off, 32));
      float mnew = fmaxf(mrun[r], tmx);
      alpha[r] = __expf(mrun[r] - mnew);
      mrun[r] = mnew;
      float ps = 0.f;
#pragma unroll
      for (int nt = 0; nt < 4; ++nt) {
        float p = __expf(sc[nt][r] - mnew);
        sc[nt][r] = p;
        ps += p;
      }
#pragma unroll
      for (int off = 1; off < 16; off <<= 1)
        ps += __shfl_xor(ps, off, 32);
      lrun[r] = lrun[r] * alpha[r] + ps;
    }

    // ---- P -> LDS (bf16) to convert C-layout to A-fragment layout; rescale O ----
#pragma unroll
    for (int nt = 0; nt < 4; ++nt) {
#pragma unroll
      for (int r = 0; r < 8; ++r) {
        Pt[wave * 16 + r + hi * 8][nt * 16 + l16] = f2bf_rne(sc[nt][r]);
        accO[nt][r] *= alpha[r];
      }
    }
    // each wave reads only rows it wrote; per-wave DS ops are in-order

    // ---- O += P V ----
#pragma unroll
    for (int ks = 0; ks < 2; ++ks) {
      const int c0 = ks * 32 + hi * 8;
      const int pr = wave * 16 + l16;
      v16bf ap = make_frag(&Pt[pr][c0], &Pt[pr][c0 + 16]);
#pragma unroll
      for (int nt = 0; nt < 4; ++nt) {
        int vr = nt * 16 + l16;                // d-row of Vs: B-frag column = d
        v16bf bv = make_frag(&Vs[buf][vr][c0], &Vs[buf][vr][c0 + 16]);
        accO[nt] = WMMA_BF16(ap, bv, accO[nt]);
      }
    }
  }

  // ---- epilogue: normalize, recombine heads into Ctx [N, D] bf16 ----
#pragma unroll
  for (int r = 0; r < 8; ++r) {
    float inv = (lrun[r] > 0.f) ? (1.0f / lrun[r]) : 0.f;
    int s_ = qb + wave * 16 + r + hi * 8;
#pragma unroll
    for (int nt = 0; nt < 4; ++nt) {
      int d_ = nt * 16 + l16;
      Ctx[((size_t)(b * S + s_)) * D + h * 64 + d_] = f2bf_rne(accO[nt][r] * inv);
    }
  }
}

// ---------------- output projection: out = Ctx @ Wo^T + bo (fp32 out) ----------------
__global__ __launch_bounds__(256) void out_gemm(
    const u16* __restrict__ Ctx, const u16* __restrict__ Wo,
    const float* __restrict__ bo, float* __restrict__ out) {
  __shared__ alignas(16) u16 As[2][128][40];
  __shared__ alignas(16) u16 Bs[2][128][40];

  const int rowBase = blockIdx.x * 128;
  const int colBase = blockIdx.y * 128;
  const int tid = threadIdx.x, lane = tid & 31, wave = tid >> 5;
  const int wm = wave & 3, wn = wave >> 2;
  const int l16 = lane & 15, hi = lane >> 4;

  auto issue_tile = [&](int buf, int k0) {
#if USE_ASYNC
#pragma unroll
    for (int i = 0; i < 2; ++i) {
      int v  = tid + i * 256;
      int r  = v >> 2;
      int c8 = (v & 3) * 8;
      async_cp16(&As[buf][r][c8], &Ctx[(size_t)(rowBase + r) * D + k0 + c8]);
      async_cp16(&Bs[buf][r][c8], &Wo[(size_t)(colBase + r) * D + k0 + c8]);
    }
#else
    uint4 ra[2], rb[2];
#pragma unroll
    for (int i = 0; i < 2; ++i) {
      int v  = tid + i * 256;
      int r  = v >> 2;
      int c8 = (v & 3) * 8;
      ra[i] = *(const uint4*)&Ctx[(size_t)(rowBase + r) * D + k0 + c8];
      rb[i] = *(const uint4*)&Wo[(size_t)(colBase + r) * D + k0 + c8];
    }
#pragma unroll
    for (int i = 0; i < 2; ++i) {
      int v  = tid + i * 256;
      int r  = v >> 2;
      int c8 = (v & 3) * 8;
      *(uint4*)&As[buf][r][c8] = ra[i];
      *(uint4*)&Bs[buf][r][c8] = rb[i];
    }
#endif
  };

  v8f acc[2][4];
  const v8f vzero = {0.f, 0.f, 0.f, 0.f, 0.f, 0.f, 0.f, 0.f};
#pragma unroll
  for (int i = 0; i < 2; ++i)
#pragma unroll
    for (int j = 0; j < 4; ++j) acc[i][j] = vzero;

  issue_tile(0, 0);
  for (int k0 = 0, j = 0; k0 < D; k0 += 32, ++j) {
    const int buf = j & 1;
    wait_async();
    __syncthreads();
    if (k0 + 32 < D) issue_tile(buf ^ 1, k0 + 32);

    const int c0 = hi * 8;
    v16bf a[2], b[4];
#pragma unroll
    for (int mt = 0; mt < 2; ++mt) {
      int r = wm * 32 + mt * 16 + l16;
      a[mt] = make_frag(&As[buf][r][c0], &As[buf][r][c0 + 16]);
    }
#pragma unroll
    for (int nt = 0; nt < 4; ++nt) {
      int r = wn * 64 + nt * 16 + l16;
      b[nt] = make_frag(&Bs[buf][r][c0], &Bs[buf][r][c0 + 16]);
    }
#pragma unroll
    for (int mt = 0; mt < 2; ++mt)
#pragma unroll
      for (int nt = 0; nt < 4; ++nt)
        acc[mt][nt] = WMMA_BF16(a[mt], b[nt], acc[mt][nt]);
  }

#pragma unroll
  for (int mt = 0; mt < 2; ++mt) {
#pragma unroll
    for (int nt = 0; nt < 4; ++nt) {
#pragma unroll
      for (int r = 0; r < 8; ++r) {
        int row = rowBase + wm * 32 + mt * 16 + r + hi * 8;
        int col = colBase + wn * 64 + nt * 16 + l16;
        out[(size_t)row * D + col] = acc[mt][nt][r] + bo[col];
      }
    }
  }
}

// ---------------- host launcher ----------------
extern "C" void kernel_launch(void* const* d_in, const int* in_sizes, int n_in,
                              void* d_out, int out_size, void* d_ws, size_t ws_size,
                              hipStream_t stream) {
  const float* query = (const float*)d_in[0];                 // [B,S,D] fp32
  const unsigned char* mask = (const unsigned char*)d_in[1];  // [B,S,S] bool (1B)
  const float* Wq = (const float*)d_in[2];
  const float* Wk = (const float*)d_in[3];
  const float* Wv = (const float*)d_in[4];
  const float* Wo = (const float*)d_in[5];
  const float* bo = (const float*)d_in[6];
  float* out = (float*)d_out;

  char* w = (char*)d_ws;
  const size_t MB = 1024ull * 1024ull;
  u16* Xbf  = (u16*)(w + 0);          //  8 MB: [N,D] bf16
  u16* Wqb  = (u16*)(w + 8  * MB);    //  2 MB each
  u16* Wkb  = (u16*)(w + 10 * MB);
  u16* Wvb  = (u16*)(w + 12 * MB);
  u16* Wob  = (u16*)(w + 14 * MB);
  u16* Qh   = (u16*)(w + 16 * MB);    //  8 MB: [B,H,S,64]
  u16* Kh   = (u16*)(w + 24 * MB);    //  8 MB: [B,H,S,64]
  u16* Vt   = (u16*)(w + 32 * MB);    //  8 MB: [B,H,64,S]
  u16* Ctx  = (u16*)(w + 40 * MB);    //  8 MB: [N,D]

  // fp32 -> bf16
  cvt_f32_bf16<<<(N * D) / 1024, 256, 0, stream>>>(query, Xbf, N * D);
  cvt_f32_bf16<<<(D * D) / 1024, 256, 0, stream>>>(Wq, Wqb, D * D);
  cvt_f32_bf16<<<(D * D) / 1024, 256, 0, stream>>>(Wk, Wkb, D * D);
  cvt_f32_bf16<<<(D * D) / 1024, 256, 0, stream>>>(Wv, Wvb, D * D);
  cvt_f32_bf16<<<(D * D) / 1024, 256, 0, stream>>>(Wo, Wob, D * D);

  // Q/K/V projections (z selects which)
  qkv_gemm<<<dim3(N / 128, D / 128, 3), 256, 0, stream>>>(Xbf, Wqb, Wkb, Wvb, Qh, Kh, Vt);

  // attention
  attn_kernel<<<dim3(S / 64, H, BB), 128, 0, stream>>>(Qh, Kh, Vt, mask, Ctx);

  // output projection + bias
  out_gemm<<<dim3(N / 128, D / 128), 256, 0, stream>>>(Ctx, Wob, bo, out);

  (void)in_sizes; (void)n_in; (void)out_size; (void)ws_size;
}
